// Model_20323785245200
// MI455X (gfx1250) — matrix operand order repaired and checked by reading, not compiled, not run on hardware
//
#include <hip/hip_runtime.h>

typedef __bf16 bf16_t;
typedef __attribute__((ext_vector_type(16))) __bf16 v16bf;
typedef __attribute__((ext_vector_type(4)))  __bf16 v4bf;
typedef __attribute__((ext_vector_type(8)))  float  v8f;

#define S_LEN 4096
#define D_DIM 1024
#define NBATCH 4
#define BM    16          // query rows per workgroup
#define NWAVES 16
#define NTHREADS 512
#define BN    256         // keys per outer block (16 waves x 16 keys)
#define SUBK  32          // keys per staged V subtile (one WMMA K-depth)
#define QK_SCALE 8.0f     // reference MULTIPLIES by sqrt(d_head)=8

// LDS row paddings chosen so row strides are odd multiples of 16B -> the 16
// lanes of an A/B fragment ds_load_b128 tile all 64 LDS banks (no conflicts).
#define SQ_LD  (D_DIM + 8)    // 2064B stride
#define SVT_LD (SUBK + 8)     // 80B stride
#define SS_LD  (BN + 4)       // 1040B stride
#define SP_LD  (BN + 8)       // 528B stride

#define USE_ASYNC_STAGE 1

union FragU { uint4 q[2]; v16bf v; };

__device__ inline void cvt4(v16bf& dst, int base, float4 f) {
    dst[base + 0] = (bf16_t)f.x;
    dst[base + 1] = (bf16_t)f.y;
    dst[base + 2] = (bf16_t)f.z;
    dst[base + 3] = (bf16_t)f.w;
}

// ---------------------------------------------------------------- prep pass
// K (f32, [b][s][d]) -> K16 (bf16, same layout)
__global__ __launch_bounds__(256)
void cvt_bf16_kernel(const float* __restrict__ src, bf16_t* __restrict__ dst,
                     int nFloat4) {
    int idx    = blockIdx.x * 256 + threadIdx.x;
    int stride = gridDim.x * 256;
    for (int i = idx; i < nFloat4; i += stride) {
        float4 f = ((const float4*)src)[i];
        v4bf o;
        o[0] = (bf16_t)f.x; o[1] = (bf16_t)f.y;
        o[2] = (bf16_t)f.z; o[3] = (bf16_t)f.w;
        ((v4bf*)dst)[i] = o;
    }
}

// V (f32, [b][s][d]) -> Vt16 (bf16, [b][d][s])  : 64x64 LDS-tiled transpose
__global__ __launch_bounds__(256)
void transpose_v_kernel(const float* __restrict__ V, bf16_t* __restrict__ Vt) {
    __shared__ bf16_t tile[64][72];        // padded rows (144B) vs bank conflicts
    const int b  = blockIdx.z;
    const int d0 = blockIdx.x * 64;
    const int k0 = blockIdx.y * 64;
    const float* Vb  = V  + ((size_t)b * S_LEN + k0) * D_DIM + d0;
    bf16_t*      Vtb = Vt + ((size_t)b * D_DIM + d0) * S_LEN + k0;
    const int tid = threadIdx.x;
    #pragma unroll
    for (int it = 0; it < 4; ++it) {
        int f4 = tid + it * 256;           // 0..1023, 16 float4 per key row
        int ky = f4 >> 4;
        int dq = (f4 & 15) * 4;
        float4 v = ((const float4*)(Vb + (size_t)ky * D_DIM))[f4 & 15];
        tile[ky][dq + 0] = (bf16_t)v.x;
        tile[ky][dq + 1] = (bf16_t)v.y;
        tile[ky][dq + 2] = (bf16_t)v.z;
        tile[ky][dq + 3] = (bf16_t)v.w;
    }
    __syncthreads();
    #pragma unroll
    for (int it = 0; it < 4; ++it) {
        int c  = tid + it * 256;           // 0..1023, 16 key-quads per d row
        int d  = c >> 4;
        int kq = (c & 15) * 4;
        v4bf o;
        o[0] = tile[kq + 0][d]; o[1] = tile[kq + 1][d];
        o[2] = tile[kq + 2][d]; o[3] = tile[kq + 3][d];
        *(v4bf*)(Vtb + (size_t)d * S_LEN + kq) = o;
    }
}

// ------------------------------------------------------------- main kernel
template<bool PREP>
__global__ __launch_bounds__(NTHREADS, 1)
void flash_attn_kernel(const float* __restrict__ Q,
                       const float* __restrict__ K,
                       const float* __restrict__ V,
                       const bf16_t* __restrict__ K16,
                       const bf16_t* __restrict__ Vt16,
                       float* __restrict__ O)
{
    __shared__ bf16_t sQ[BM][SQ_LD];        // ~32KB  Q tile (bf16)
    __shared__ bf16_t sVt[D_DIM][SVT_LD];   // ~80KB  transposed V subtile [d][key]
    __shared__ float  sS[BM][SS_LD];        // ~16KB  fp32 scores (scaled)
    __shared__ bf16_t sP[BM][SP_LD];        // ~8KB   bf16 probabilities
    __shared__ float  sM[BM], sL[BM], sC[BM];

    const int tid   = threadIdx.x;
    const int wv    = tid >> 5;             // wave id 0..15
    const int lane  = tid & 31;
    const int laneM = lane & 15;            // row (A) / column (B) index
    const int hi    = lane >> 4;            // lane half selects K-chunk group

    const int blk = blockIdx.x;             // 1024 blocks = 4 batches * 256 q-tiles
    const int b   = blk >> 8;
    const int q0  = (blk & 255) * BM;

    const size_t batchOff = (size_t)b * S_LEN * D_DIM;
    const float*  Qb   = Q + batchOff + (size_t)q0 * D_DIM;
    const float*  Kb   = K + batchOff;
    const float*  Vb   = V + batchOff;
    const bf16_t* Kb16 = K16 + batchOff;
    const bf16_t* Vtb  = Vt16 + batchOff;   // [d][s] layout
    float*        Ob   = O + batchOff + (size_t)q0 * D_DIM;

    if (tid < BM) { sM[tid] = -3.0e38f; sL[tid] = 0.0f; }

    // ---- stage Q tile (16x1024 f32 -> bf16 in LDS), coalesced float4 reads
    #pragma unroll
    for (int it = 0; it < 8; ++it) {
        int f4  = tid + it * NTHREADS;      // 0..4095 (256 float4 per row)
        int row = f4 >> 8;
        int col = (f4 & 255) * 4;
        float4 v = ((const float4*)(Qb + (size_t)row * D_DIM))[f4 & 255];
        sQ[row][col + 0] = (bf16_t)v.x;
        sQ[row][col + 1] = (bf16_t)v.y;
        sQ[row][col + 2] = (bf16_t)v.z;
        sQ[row][col + 3] = (bf16_t)v.w;
    }
    __syncthreads();

    v8f acc[4];                             // wave's 16x64 output slice (f32)
    #pragma unroll
    for (int t = 0; t < 4; ++t) acc[t] = (v8f){};

    for (int kb = 0; kb < S_LEN / BN; ++kb) {
        // ============ scores: wave wv -> 16x16 tile, K-depth 1024 ==========
        const int keyBase = kb * BN + wv * 16;
        const float*  kRow   = Kb   + (size_t)(keyBase + laneM) * D_DIM;
        const bf16_t* kRow16 = Kb16 + (size_t)(keyBase + laneM) * D_DIM;
        v8f sacc = (v8f){};
        #pragma unroll 4
        for (int kc = 0; kc < D_DIM; kc += 32) {
            // A fragment (Q): lane holds 16 contiguous K values {c0..+15}
            const int c0 = kc + hi * 16;
            FragU au;
            au.q[0] = *(const uint4*)&sQ[laneM][c0];
            au.q[1] = *(const uint4*)&sQ[laneM][c0 + 8];
            // B fragment (K^T): lane holds 16 contiguous d-values at kc+hi*16
            FragU bu;
            if constexpr (PREP) {
                const bf16_t* kp = kRow16 + kc + hi * 16;  // 32B contiguous
                bu.q[0] = ((const uint4*)kp)[0];
                bu.q[1] = ((const uint4*)kp)[1];
            } else {
                const float* kp = kRow + kc + hi * 16;
                cvt4(bu.v, 0,  ((const float4*)kp)[0]);
                cvt4(bu.v, 4,  ((const float4*)kp)[1]);
                cvt4(bu.v, 8,  ((const float4*)kp)[2]);
                cvt4(bu.v, 12, ((const float4*)kp)[3]);
            }
            sacc = __builtin_amdgcn_wmma_f32_16x16x32_bf16(
                       false, au.v, false, bu.v, (short)0, sacc, false, false);
        }
        // C layout: VGPR i -> row i (lanes 0-15) / row i+8 (lanes 16-31)
        #pragma unroll
        for (int i = 0; i < 8; ++i)
            sS[i + hi * 8][wv * 16 + laneM] = sacc[i] * QK_SCALE;
        __syncthreads();

        // ============ online softmax: wave r handles row r ================
        {
            const int r = wv;
            float v0[8];
            float mx = -3.0e38f;
            #pragma unroll
            for (int j = 0; j < 8; ++j) {
                v0[j] = sS[r][lane + 32 * j];
                mx = fmaxf(mx, v0[j]);
            }
            #pragma unroll
            for (int off = 16; off >= 1; off >>= 1)
                mx = fmaxf(mx, __shfl_xor(mx, off, 32));
            const float mOld = sM[r];
            const float mNew = fmaxf(mOld, mx);
            float sum = 0.0f;
            #pragma unroll
            for (int j = 0; j < 8; ++j) {
                float p = __expf(v0[j] - mNew);
                sum += p;
                sP[r][lane + 32 * j] = (bf16_t)p;
            }
            #pragma unroll
            for (int off = 16; off >= 1; off >>= 1)
                sum += __shfl_xor(sum, off, 32);
            if (lane == 0) {
                float corr = __expf(mOld - mNew);
                sC[r] = corr;
                sM[r] = mNew;
                sL[r] = sL[r] * corr + sum;
            }
        }
        __syncthreads();

        // rescale running accumulator by per-row correction factor
        #pragma unroll
        for (int i = 0; i < 8; ++i) {
            const float c = sC[i + hi * 8];
            #pragma unroll
            for (int t = 0; t < 4; ++t) acc[t][i] *= c;
        }

        // ============ P @ V over 8 subtiles of 32 keys ====================
        for (int sub = 0; sub < BN / SUBK; ++sub) {
            __syncthreads();                // protect sVt from prior readers
            const int ks = kb * BN + sub * SUBK;
            if constexpr (PREP) {
                // Vt16 is already bf16 [d][s]: contiguous 16B copies to LDS.
                #pragma unroll
                for (int it = 0; it < 8; ++it) {
                    int chunk = tid + it * NTHREADS;   // 0..4095
                    int d  = chunk >> 2;
                    int p8 = (chunk & 3) * 8;
                    const bf16_t* gp = Vtb + (size_t)d * S_LEN + ks + p8;
#if USE_ASYNC_STAGE
                    unsigned lds_off = (unsigned)(uintptr_t)&sVt[d][p8];
                    asm volatile("global_load_async_to_lds_b128 %0, %1, off"
                                 :: "v"(lds_off), "v"(gp) : "memory");
#else
                    *(uint4*)&sVt[d][p8] = *(const uint4*)gp;
#endif
                }
#if USE_ASYNC_STAGE
                asm volatile("s_wait_asynccnt 0" ::: "memory");
#endif
            } else {
                // fallback: transpose f32 V in-kernel, element-wise
                #pragma unroll
                for (int it = 0; it < 16; ++it) {
                    int f4  = tid + it * NTHREADS;     // 0..8191
                    int key = f4 >> 8;
                    int dq  = (f4 & 255) * 4;
                    float4 v = ((const float4*)(Vb + (size_t)(ks + key) * D_DIM))[f4 & 255];
                    sVt[dq + 0][key] = (bf16_t)v.x;
                    sVt[dq + 1][key] = (bf16_t)v.y;
                    sVt[dq + 2][key] = (bf16_t)v.z;
                    sVt[dq + 3][key] = (bf16_t)v.w;
                }
            }
            __syncthreads();
            // A fragment (P): rows 0..15, K = keys of this subtile
            const int pc0 = sub * SUBK + hi * 16;
            FragU pu;
            pu.q[0] = *(const uint4*)&sP[laneM][pc0];
            pu.q[1] = *(const uint4*)&sP[laneM][pc0 + 8];
            #pragma unroll
            for (int t = 0; t < 4; ++t) {
                const int d = wv * 64 + t * 16 + laneM;  // B column = d index
                FragU bu;
                bu.q[0] = *(const uint4*)&sVt[d][hi * 16];
                bu.q[1] = *(const uint4*)&sVt[d][hi * 16 + 8];
                acc[t] = __builtin_amdgcn_wmma_f32_16x16x32_bf16(
                             false, pu.v, false, bu.v, (short)0, acc[t], false, false);
            }
        }
    }

    // ============ finalize: divide by running sum, store fp32 =============
    #pragma unroll
    for (int i = 0; i < 8; ++i) {
        const int row   = i + hi * 8;
        const float inv = 1.0f / sL[row];
        float* orow = Ob + (size_t)row * D_DIM;
        #pragma unroll
        for (int t = 0; t < 4; ++t)
            orow[wv * 64 + t * 16 + laneM] = acc[t][i] * inv;
    }
}

extern "C" void kernel_launch(void* const* d_in, const int* in_sizes, int n_in,
                              void* d_out, int out_size, void* d_ws, size_t ws_size,
                              hipStream_t stream) {
    (void)in_sizes; (void)n_in; (void)out_size;
    const float* Q = (const float*)d_in[0];
    const float* K = (const float*)d_in[1];
    const float* V = (const float*)d_in[2];
    float*       O = (float*)d_out;

    const size_t nElem = (size_t)NBATCH * S_LEN * D_DIM;       // 16.7M
    const size_t need  = 2 * nElem * sizeof(bf16_t);           // 64MB: K16 + Vt16

    dim3 grid(NBATCH * (S_LEN / BM));   // 1024 workgroups
    dim3 block(NTHREADS);

    if (d_ws != nullptr && ws_size >= need) {
        bf16_t* K16  = (bf16_t*)d_ws;
        bf16_t* Vt16 = K16 + nElem;
        cvt_bf16_kernel<<<2048, 256, 0, stream>>>(K, K16, (int)(nElem / 4));
        transpose_v_kernel<<<dim3(D_DIM / 64, S_LEN / 64, NBATCH), 256, 0, stream>>>(V, Vt16);
        flash_attn_kernel<true><<<grid, block, 0, stream>>>(Q, K, V, K16, Vt16, O);
    } else {
        flash_attn_kernel<false><<<grid, block, 0, stream>>>(Q, K, V, nullptr, nullptr, O);
    }
}
